// Efficient_ViG_blocks_1632087572834
// MI455X (gfx1250) — matrix operand order repaired and checked by reading, not compiled, not run on hardware
//
#include <hip/hip_runtime.h>
#include <hip/hip_bf16.h>
#include <stdint.h>

// ---------------------------------------------------------------------------
// Efficient-ViG block for MI455X (gfx1250, wave32, WMMA).
// All GEMMs run on v_wmma_f32_16x16x32_bf16 (f32 accumulate, BN folded into
// bf16 weights). Activations are pixel-major so WMMA fragments are contiguous
// dword loads and C-tiles store 8 contiguous channels per lane.
// GEMM waves own a 16(ch) x 64(px) tile: 1 weight fragment reused across 4
// independent accumulators. Gram+topk waves process 2 key tiles/iter with 2
// independent accumulator chains sharing resident query fragments.
// ---------------------------------------------------------------------------

typedef __bf16 v16bf __attribute__((ext_vector_type(16)));
typedef float  v8f   __attribute__((ext_vector_type(8)));
struct Frag32 { uint32_t d[8]; };

__device__ __forceinline__ uint16_t f2bf(float x) {
  uint32_t u = __builtin_bit_cast(uint32_t, x);
  uint32_t r = u + 0x7fffu + ((u >> 16) & 1u);   // round-to-nearest-even
  return (uint16_t)(r >> 16);
}
__device__ __forceinline__ float bf2f(uint16_t h) {
  uint32_t u = ((uint32_t)h) << 16;
  return __builtin_bit_cast(float, u);
}

// ------------------------- BN-fold weights into bf16 ------------------------
__global__ void k_fold(const float* __restrict__ w, const float* __restrict__ cb,
                       const float* __restrict__ g, const float* __restrict__ be,
                       const float* __restrict__ m, const float* __restrict__ v,
                       uint16_t* __restrict__ wOut, float* __restrict__ tOut,
                       int Cout, int Cin) {
  int t = blockIdx.x * 256 + threadIdx.x;
  int tot = Cout * Cin;
  if (t < tot) {
    int o = t / Cin;
    float s = g[o] * rsqrtf(v[o] + 1e-5f);
    wOut[t] = f2bf(s * w[t]);
  }
  if (t < Cout) {
    float s = g[t] * rsqrtf(v[t] + 1e-5f);
    tOut[t] = (cb[t] - m[t]) * s + be[t];
  }
}

// -------------------- transpose x (B,C,HW) -> xT (B,HW,C) bf16 --------------
__global__ __launch_bounds__(256) void k_transpose_x(const float* __restrict__ x,
                                                     uint16_t* __restrict__ xT,
                                                     int C, int HW) {
  __shared__ uint16_t tile[32][33];
  int b = blockIdx.z;
  int cbase = blockIdx.y * 32, pbase = blockIdx.x * 32;
  const float* xb = x + (size_t)b * C * HW;
  #pragma unroll
  for (int i = 0; i < 4; ++i) {
    int c = cbase + threadIdx.y + i * 8;
    tile[threadIdx.y + i * 8][threadIdx.x] =
        f2bf(xb[(size_t)c * HW + pbase + threadIdx.x]);
  }
  __syncthreads();
  uint16_t* xo = xT + (size_t)b * HW * C;
  #pragma unroll
  for (int i = 0; i < 4; ++i) {
    int p = pbase + threadIdx.y + i * 8;
    xo[(size_t)p * C + cbase + threadIdx.x] = tile[threadIdx.x][threadIdx.y + i * 8];
  }
}

// ------------------------------ generic WMMA GEMM ---------------------------
// out[p, ch] = relu?( act(NPxCin,bf16) @ w'(CoutxCin,bf16)^T + bias ) + residual
// One wave computes a 16(ch) x 64(px) tile (4 accumulators); 8 waves / block.
__global__ __launch_bounds__(256) void k_gemm(
    const uint16_t* __restrict__ act, const uint16_t* __restrict__ w,
    const float* __restrict__ bias, int NP, int Cin, int Cout, int relu,
    const float* __restrict__ res_pm,   // pixel-major residual (or null)
    const float* __restrict__ res_cm,   // channel-major residual (or null)
    float* __restrict__ out_pm, uint16_t* __restrict__ out_bf,
    float* __restrict__ out_cm) {
  const int b = blockIdx.y;
  const int lane = threadIdx.x & 31, wv = threadIdx.x >> 5;
  const int lm = lane & 15, half = lane >> 4;
  const int ctiles = Cout >> 4;
  const int tiles = (NP >> 6) * ctiles;          // 64 pixels per wave-tile
  const int tile = blockIdx.x * 8 + wv;
  if (tile >= tiles) return;
  const int ptile = (tile / ctiles) << 6;
  const int ctile = (tile % ctiles) << 4;

  const uint32_t* a32 = (const uint32_t*)w;
  const uint32_t* b32 = (const uint32_t*)act + (((size_t)b * NP * Cin) >> 1);
  const int arow = ((ctile + lm) * Cin) >> 1;
  int brow[4];
  #pragma unroll
  for (int j = 0; j < 4; ++j) brow[j] = ((ptile + 16 * j + lm) * Cin) >> 1;

  v8f acc0 = {0.f,0.f,0.f,0.f,0.f,0.f,0.f,0.f};
  v8f acc1 = acc0, acc2 = acc0, acc3 = acc0;

  for (int ko = 0; ko < Cin; ko += 32) {
    Frag32 af, bf0, bf1, bf2, bf3;
    #pragma unroll
    for (int v2 = 0; v2 < 8; ++v2) {
      // ISA 16-bit A (16x32) layout:  k = (v<4 ? 2v : 2v+8) + 8*half
      int ka = ko + ((v2 < 4) ? (2 * v2) : (2 * v2 + 8)) + 8 * half;
      af.d[v2] = a32[arow + (ka >> 1)];
      // ISA 16-bit B (32x16) layout:  k = (v<4 ? 2v : 2v+8) + 8*half
      int kb = ko + ((v2 < 4) ? (2 * v2) : (2 * v2 + 8)) + 8 * half;
      bf0.d[v2] = b32[brow[0] + (kb >> 1)];
      bf1.d[v2] = b32[brow[1] + (kb >> 1)];
      bf2.d[v2] = b32[brow[2] + (kb >> 1)];
      bf3.d[v2] = b32[brow[3] + (kb >> 1)];
    }
    if (ko + 32 < Cin) {
      __builtin_prefetch(&b32[brow[0] + ((ko + 32) >> 1)], 0, 1);
      __builtin_prefetch(&b32[brow[2] + ((ko + 32) >> 1)], 0, 1);
    }
    v16bf A  = __builtin_bit_cast(v16bf, af);
    v16bf B0 = __builtin_bit_cast(v16bf, bf0);
    v16bf B1 = __builtin_bit_cast(v16bf, bf1);
    v16bf B2 = __builtin_bit_cast(v16bf, bf2);
    v16bf B3 = __builtin_bit_cast(v16bf, bf3);
    acc0 = __builtin_amdgcn_wmma_f32_16x16x32_bf16(false, A, false, B0, (short)0, acc0, false, false);
    acc1 = __builtin_amdgcn_wmma_f32_16x16x32_bf16(false, A, false, B1, (short)0, acc1, false, false);
    acc2 = __builtin_amdgcn_wmma_f32_16x16x32_bf16(false, A, false, B2, (short)0, acc2, false, false);
    acc3 = __builtin_amdgcn_wmma_f32_16x16x32_bf16(false, A, false, B3, (short)0, acc3, false, false);
  }

  const int ch0 = ctile + half * 8;   // C layout: vgpr r -> M = 8*half + r
  float bb[8];
  #pragma unroll
  for (int r = 0; r < 8; ++r) bb[r] = bias[ch0 + r];

  #pragma unroll
  for (int j = 0; j < 4; ++j) {
    const v8f& acc = (j == 0) ? acc0 : (j == 1) ? acc1 : (j == 2) ? acc2 : acc3;
    const int p = ptile + 16 * j + lm;
    float vout[8];
    #pragma unroll
    for (int r = 0; r < 8; ++r) {
      float vv = acc[r] + bb[r];
      if (relu) vv = fmaxf(vv, 0.f);
      vout[r] = vv;
    }
    if (res_pm) {
      const float* rp = res_pm + (size_t)b * NP * Cout + (size_t)p * Cout + ch0;
      #pragma unroll
      for (int r = 0; r < 8; ++r) vout[r] += rp[r];
    }
    if (res_cm) {
      const float* rc = res_cm + (size_t)b * Cout * NP + p;
      #pragma unroll
      for (int r = 0; r < 8; ++r) vout[r] += rc[(size_t)(ch0 + r) * NP];
    }
    if (out_pm) {
      float* o = out_pm + (size_t)b * NP * Cout + (size_t)p * Cout + ch0;
      #pragma unroll
      for (int r = 0; r < 8; ++r) o[r] = vout[r];
    }
    if (out_bf) {
      uint16_t* o = out_bf + (size_t)b * NP * Cout + (size_t)p * Cout + ch0;
      #pragma unroll
      for (int r = 0; r < 8; ++r) o[r] = f2bf(vout[r]);
    }
    if (out_cm) {
      float* o = out_cm + (size_t)b * Cout * NP + p;
      #pragma unroll
      for (int r = 0; r < 8; ++r) o[(size_t)(ch0 + r) * NP] = vout[r];
    }
  }
}

// ------------------------- 2x2 maxpool (+argmax, u8) ------------------------
__global__ void k_pool(const uint16_t* __restrict__ h, uint16_t* __restrict__ hp,
                       uint8_t* __restrict__ idx, int Hp, int Wp, int C, int W) {
  int b = blockIdx.y;
  int t = blockIdx.x * 256 + threadIdx.x;
  int N = Hp * Wp;
  if (t >= N * C) return;
  int n = t / C, c = t - n * C;
  int pr = n / Wp, pc = n - pr * Wp;
  const uint16_t* hb = h + (size_t)b * (size_t)(2 * Hp) * (2 * Wp) * C;
  float best = -3e38f; int boff = 0;
  #pragma unroll
  for (int dh = 0; dh < 2; ++dh)
    #pragma unroll
    for (int dw = 0; dw < 2; ++dw) {
      int p = (2 * pr + dh) * W + 2 * pc + dw;
      float v = bf2f(hb[(size_t)p * C + c]);
      if (v > best) { best = v; boff = dh * 2 + dw; }
    }
  hp[((size_t)b * N + n) * C + c] = f2bf(best);
  idx[((size_t)b * N + n) * C + c] = (uint8_t)boff;
}

// ----------------- per-point L2 normalize (one wave / point) ----------------
__global__ __launch_bounds__(256) void k_norm(const uint16_t* __restrict__ hp,
                                              uint16_t* __restrict__ xn,
                                              int N, int C) {
  int b = blockIdx.y;
  int wv = threadIdx.x >> 5, lane = threadIdx.x & 31;
  int pt = blockIdx.x * 8 + wv;
  if (pt >= N) return;
  const uint16_t* src = hp + ((size_t)b * N + pt) * C;
  float v0 = bf2f(src[lane]), v1 = bf2f(src[lane + 32]), v2 = bf2f(src[lane + 64]);
  float s = v0 * v0 + v1 * v1 + v2 * v2;
  #pragma unroll
  for (int o = 16; o > 0; o >>= 1) s += __shfl_xor(s, o, 32);
  float rn = 1.f / (sqrtf(s) + 1e-12f);
  uint16_t* dst = xn + ((size_t)b * N + pt) * C;
  dst[lane] = f2bf(v0 * rn);
  dst[lane + 32] = f2bf(v1 * rn);
  dst[lane + 64] = f2bf(v2 * rn);
}

// --------- fused gram (WMMA bf16) + streaming top-9 neighbor search ---------
// One wave owns 16 query points (columns N of the WMMA tile); loops over key
// tiles 2-at-a-time (2 independent accumulator chains, shared query operand).
// Per-lane sorted top-9, lanes l / l+16 merged via shfl at the end.
__global__ __launch_bounds__(256) void k_gram_topk(const uint16_t* __restrict__ xn,
                                                   int* __restrict__ nbr,
                                                   int N, int C) {
  const int b = blockIdx.y;
  const int lane = threadIdx.x & 31, wv = threadIdx.x >> 5;
  const int lm = lane & 15, half = lane >> 4;
  const int q0 = (blockIdx.x * 8 + wv) << 4;
  if (q0 >= N) return;
  const uint32_t* x32 = (const uint32_t*)xn + (((size_t)b * N * C) >> 1);

  Frag32 qf[3];                       // query fragments (B operand), C=96 -> 3 chunks
  const int qrow = ((q0 + lm) * C) >> 1;
  #pragma unroll
  for (int s = 0; s < 3; ++s)
    #pragma unroll
    for (int v2 = 0; v2 < 8; ++v2) {
      int kb = 32 * s + ((v2 < 4) ? (2 * v2) : (2 * v2 + 8)) + 8 * half;
      qf[s].d[v2] = x32[qrow + (kb >> 1)];
    }

  float tv[9]; int ti[9];
  #pragma unroll
  for (int t = 0; t < 9; ++t) { tv[t] = -3e38f; ti[t] = 0; }

  const int ntiles = N >> 4;          // 576: even -> exact 2-wide loop
  for (int kt = 0; kt < ntiles; kt += 2) {
    const int k0 = kt << 4;
    const int krow0 = ((k0 + lm) * C) >> 1;
    const int krow1 = ((k0 + 16 + lm) * C) >> 1;
    v8f acc0 = {0.f, 0.f, 0.f, 0.f, 0.f, 0.f, 0.f, 0.f};
    v8f acc1 = acc0;
    #pragma unroll
    for (int s = 0; s < 3; ++s) {
      Frag32 af0, af1;                // key fragments (A operand)
      #pragma unroll
      for (int v2 = 0; v2 < 8; ++v2) {
        int ka = 32 * s + ((v2 < 4) ? (2 * v2) : (2 * v2 + 8)) + 8 * half;
        af0.d[v2] = x32[krow0 + (ka >> 1)];
        af1.d[v2] = x32[krow1 + (ka >> 1)];
      }
      v16bf A0 = __builtin_bit_cast(v16bf, af0);
      v16bf A1 = __builtin_bit_cast(v16bf, af1);
      v16bf B  = __builtin_bit_cast(v16bf, qf[s]);
      acc0 = __builtin_amdgcn_wmma_f32_16x16x32_bf16(false, A0, false, B,
                                                     (short)0, acc0, false, false);
      acc1 = __builtin_amdgcn_wmma_f32_16x16x32_bf16(false, A1, false, B,
                                                     (short)0, acc1, false, false);
    }
    #pragma unroll
    for (int u = 0; u < 2; ++u) {
      const v8f& acc = (u == 0) ? acc0 : acc1;
      const int kbase = k0 + 16 * u + half * 8;
      #pragma unroll
      for (int r = 0; r < 8; ++r) {
        float v = acc[r];
        int ki = kbase + r;
        if (v > tv[8]) {              // insertion into sorted-desc top-9
          int j = 8;
          while (j > 0 && tv[j - 1] < v) { tv[j] = tv[j - 1]; ti[j] = ti[j - 1]; --j; }
          tv[j] = v; ti[j] = ki;
        }
      }
    }
  }

  // merge lane l (keys m=0..7 of each tile) with lane l+16 (m=8..15)
  float pv[9]; int pi[9];
  #pragma unroll
  for (int t = 0; t < 9; ++t) {
    pv[t] = __shfl_xor(tv[t], 16, 32);
    pi[t] = __shfl_xor(ti[t], 16, 32);
  }
  int mi[9]; int i = 0, j = 0;
  #pragma unroll
  for (int t = 0; t < 9; ++t) {
    bool ta = (tv[i] > pv[j]) || (tv[i] == pv[j] && ti[i] < pi[j]);
    mi[t] = ta ? ti[i] : pi[j];
    i += ta ? 1 : 0; j += ta ? 0 : 1;
  }
  if (half == 0) {
    int* o = nbr + ((size_t)b * N + q0 + lm) * 9;
    #pragma unroll
    for (int t = 0; t < 9; ++t) o[t] = mi[t];
  }
}

// ----- build graph features s[n, 2c]=x_c, s[n, 2c+1]=max_j x_j,c - x_c ------
__global__ void k_build_s(const uint16_t* __restrict__ hp, const int* __restrict__ nbr,
                          uint16_t* __restrict__ s, int N, int C) {
  int b = blockIdx.y;
  int t = blockIdx.x * 256 + threadIdx.x;
  if (t >= N * C) return;
  int n = t / C, c = t - n * C;
  const uint16_t* hb = hp + (size_t)b * N * C;
  float xi = bf2f(hb[(size_t)n * C + c]);
  const int* nb = nbr + ((size_t)b * N + n) * 9;
  float m = -3e38f;
  #pragma unroll
  for (int k = 0; k < 9; ++k) m = fmaxf(m, bf2f(hb[(size_t)nb[k] * C + c]));
  uint16_t* so = s + ((size_t)b * N + n) * (2 * C);
  so[2 * c] = f2bf(xi);
  so[2 * c + 1] = f2bf(m - xi);
}

// ------------------------------- 2x2 maxunpool ------------------------------
__global__ void k_unpool(const uint16_t* __restrict__ g, const uint8_t* __restrict__ idx,
                         uint16_t* __restrict__ gf, int H, int W, int C, int C2) {
  int b = blockIdx.y;
  long long t = (long long)blockIdx.x * 256 + threadIdx.x;
  long long tot = (long long)H * W * C2;
  if (t >= tot) return;
  int p = (int)(t / C2); int c2 = (int)(t - (long long)p * C2);
  int r = p / W, cl = p - r * W;
  int Wp = W >> 1;
  int n = (r >> 1) * Wp + (cl >> 1);
  int off = (r & 1) * 2 + (cl & 1);
  int cref = (c2 < C) ? c2 : (c2 - C);
  int N = (H >> 1) * Wp;
  uint16_t val = 0;
  if (idx[((size_t)b * N + n) * C + cref] == (uint8_t)off)
    val = g[((size_t)b * N + n) * C2 + c2];
  gf[((size_t)b * H * W + p) * C2 + c2] = val;
}

// ---------------------------------------------------------------------------
extern "C" void kernel_launch(void* const* d_in, const int* in_sizes, int n_in,
                              void* d_out, int out_size, void* d_ws, size_t ws_size,
                              hipStream_t stream) {
  (void)in_sizes; (void)n_in; (void)out_size; (void)ws_size;
  const int B = 2, C = 96, H = 192, W = 192, HW = H * W;
  const int Hp = 96, Wp = 96, N = Hp * Wp, C2 = 192, CH = 384;

  const float* x     = (const float*)d_in[0];
  const float* fc1_w = (const float*)d_in[1];  const float* fc1_b = (const float*)d_in[2];
  const float* bn1g  = (const float*)d_in[3];  const float* bn1b  = (const float*)d_in[4];
  const float* bn1m  = (const float*)d_in[5];  const float* bn1v  = (const float*)d_in[6];
  const float* g_w   = (const float*)d_in[7];  const float* g_b   = (const float*)d_in[8];
  const float* bngg  = (const float*)d_in[9];  const float* bngb  = (const float*)d_in[10];
  const float* bngm  = (const float*)d_in[11]; const float* bngv  = (const float*)d_in[12];
  const float* fc2_w = (const float*)d_in[13]; const float* fc2_b = (const float*)d_in[14];
  const float* bn2g  = (const float*)d_in[15]; const float* bn2b  = (const float*)d_in[16];
  const float* bn2m  = (const float*)d_in[17]; const float* bn2v  = (const float*)d_in[18];
  const float* ff1_w = (const float*)d_in[19]; const float* ff1_b = (const float*)d_in[20];
  const float* bf1g  = (const float*)d_in[21]; const float* bf1b  = (const float*)d_in[22];
  const float* bf1m  = (const float*)d_in[23]; const float* bf1v  = (const float*)d_in[24];
  const float* ff2_w = (const float*)d_in[25]; const float* ff2_b = (const float*)d_in[26];
  const float* bf2g  = (const float*)d_in[27]; const float* bf2b  = (const float*)d_in[28];
  const float* bf2m  = (const float*)d_in[29]; const float* bf2v  = (const float*)d_in[30];

  uint8_t* ws = (uint8_t*)d_ws;
  const size_t MB = 1ull << 20;
  // folded weights / biases
  uint16_t* W1  = (uint16_t*)(ws + 0);
  uint16_t* WG  = (uint16_t*)(ws + 32768);
  uint16_t* W2  = (uint16_t*)(ws + 131072);
  uint16_t* WF1 = (uint16_t*)(ws + 196608);
  uint16_t* WF2 = (uint16_t*)(ws + 327680);
  float* T1  = (float*)(ws + 450560);
  float* TG  = (float*)(ws + 452608);
  float* T2  = (float*)(ws + 454656);
  float* TF1 = (float*)(ws + 456704);
  float* TF2 = (float*)(ws + 458752);
  // activations (pixel-major). GF aliases XT+HB; F aliases HP..G (all dead).
  uint16_t* XT  = (uint16_t*)(ws + 1 * MB);     // B*HW*C bf16 (14.2 MB)
  uint16_t* HB  = (uint16_t*)(ws + 16 * MB);    // B*HW*C bf16 (14.2 MB)
  uint16_t* GF  = (uint16_t*)(ws + 1 * MB);     // B*HW*C2 bf16 (28.3 MB) alias
  uint16_t* HP  = (uint16_t*)(ws + 31 * MB);    // B*N*C bf16
  uint8_t*  IDX = (uint8_t*)(ws + 35 * MB);     // B*N*C u8
  uint16_t* XN  = (uint16_t*)(ws + 37 * MB);    // B*N*C bf16
  int*      NBR = (int*)(ws + 41 * MB);         // B*N*9 i32
  uint16_t* S   = (uint16_t*)(ws + 42 * MB);    // B*N*C2 bf16
  uint16_t* G   = (uint16_t*)(ws + 50 * MB);    // B*N*C2 bf16
  uint16_t* F   = (uint16_t*)(ws + 31 * MB);    // B*HW*CH bf16 (54 MB) alias
  float*    H2  = (float*)(ws + 88 * MB);       // B*HW*C f32 (28.3 MB)
  uint16_t* H2B = (uint16_t*)(ws + 116 * MB);   // B*HW*C bf16 (14.2 MB)

  // 1) fold BN into weights (bf16) and biases (f32)
  k_fold<<<dim3((C * C + 255) / 256), 256, 0, stream>>>(fc1_w, fc1_b, bn1g, bn1b, bn1m, bn1v, W1, T1, C, C);
  k_fold<<<dim3((C2 * C2 + 255) / 256), 256, 0, stream>>>(g_w, g_b, bngg, bngb, bngm, bngv, WG, TG, C2, C2);
  k_fold<<<dim3((C * C2 + 255) / 256), 256, 0, stream>>>(fc2_w, fc2_b, bn2g, bn2b, bn2m, bn2v, W2, T2, C, C2);
  k_fold<<<dim3((CH * C + 255) / 256), 256, 0, stream>>>(ff1_w, ff1_b, bf1g, bf1b, bf1m, bf1v, WF1, TF1, CH, C);
  k_fold<<<dim3((C * CH + 255) / 256), 256, 0, stream>>>(ff2_w, ff2_b, bf2g, bf2b, bf2m, bf2v, WF2, TF2, C, CH);

  // 2) transpose x to pixel-major bf16
  k_transpose_x<<<dim3(HW / 32, C / 32, B), dim3(32, 8), 0, stream>>>(x, XT, C, HW);

  // 3) h = bn1(fc1(x))  -> HB (bf16, pixel-major)
  k_gemm<<<dim3(((HW / 64) * (C / 16)) / 8, B), 256, 0, stream>>>(
      XT, W1, T1, HW, C, C, 0, nullptr, nullptr, nullptr, HB, nullptr);

  // 4) maxpool 2x2 + argmax
  k_pool<<<dim3((N * C + 255) / 256, B), 256, 0, stream>>>(HB, HP, IDX, Hp, Wp, C, W);

  // 5) L2 normalize points
  k_norm<<<dim3(N / 8, B), 256, 0, stream>>>(HP, XN, N, C);

  // 6) fused gram + top-9 KNN
  k_gram_topk<<<dim3(N / 128, B), 256, 0, stream>>>(XN, NBR, N, C);

  // 7) graph features s
  k_build_s<<<dim3((N * C + 255) / 256, B), 256, 0, stream>>>(HP, NBR, S, N, C);

  // 8) g = relu(bng(conv_g(s)))
  k_gemm<<<dim3(((N / 64) * (C2 / 16)) / 8, B), 256, 0, stream>>>(
      S, WG, TG, N, C2, C2, 1, nullptr, nullptr, nullptr, G, nullptr);

  // 9) maxunpool to full resolution
  k_unpool<<<dim3((HW * C2 + 255) / 256, B), 256, 0, stream>>>(G, IDX, GF, H, W, C, C2);

  // 10) h2 = bn2(fc2(g_full)) + x   (f32 + bf16 copies)
  k_gemm<<<dim3(((HW / 64) * (C / 16)) / 8, B), 256, 0, stream>>>(
      GF, W2, T2, HW, C2, C, 0, nullptr, x, H2, H2B, nullptr);

  // 11) f = relu(bnf1(ff1(h2)))
  k_gemm<<<dim3(((HW / 64) * (CH / 16)) / 8, B), 256, 0, stream>>>(
      H2B, WF1, TF1, HW, C, CH, 1, nullptr, nullptr, nullptr, F, nullptr);

  // 12) out = bnf2(ff2(f)) + h2   (channel-major f32 to d_out)
  k_gemm<<<dim3(((HW / 64) * (C / 16)) / 8, B), 256, 0, stream>>>(
      F, WF2, TF2, HW, CH, C, 0, H2, nullptr, nullptr, nullptr, (float*)d_out);
}
